// m_SCNN_72705206386765
// MI455X (gfx1250) — compile-verified, matrix operand order checked by reading, not run on hardware
//
#include <hip/hip_runtime.h>

typedef __attribute__((ext_vector_type(16))) __bf16 v16bf;
typedef __attribute__((ext_vector_type(8)))  __bf16 v8bf;
typedef __attribute__((ext_vector_type(8)))  float  v8f;
typedef __attribute__((ext_vector_type(4)))  float  v4f;

#define CCH 128   // channels
#define HH  128   // height
#define WW  192   // width
#define KK  9     // conv taps
#define PAD 4     // KK/2

#define LDS_STRIDE 144   // bf16 elems per carry row (128 ic + pad; 288B rows, 32B aligned)
#define LDS_ROWS   200   // max(192,128) + 2*PAD

// ---------------------------------------------------------------------------
// Convert fp32 weights [oc][ic][t] -> bf16 [pass][t][oc][ic] in workspace
// ---------------------------------------------------------------------------
__global__ __launch_bounds__(512)
void convert_weights_kernel(const float* __restrict__ w0, const float* __restrict__ w1,
                            const float* __restrict__ w2, const float* __restrict__ w3,
                            __bf16* __restrict__ wb)
{
    const float* srcs[4] = {w0, w1, w2, w3};
    const int total = CCH * CCH * KK;
    for (int idx = blockIdx.x * blockDim.x + threadIdx.x; idx < total;
         idx += gridDim.x * blockDim.x) {
        int t  = idx % KK;
        int ic = (idx / KK) % CCH;
        int oc = idx / (KK * CCH);
        size_t dst = ((size_t)t * CCH + oc) * CCH + ic;
        #pragma unroll
        for (int p = 0; p < 4; ++p) {
            wb[(size_t)p * KK * CCH * CCH + dst] = (__bf16)srcs[p][idx];
        }
    }
}

// ---------------------------------------------------------------------------
// Async global -> LDS copy of one scan row r into ldsX ([c][pos] fp32 layout).
// VERT rows are contiguous per channel -> b128 chunks; HORZ is stride-W -> b32.
// Tracked by ASYNCcnt; caller must s_wait_asynccnt + barrier before reading.
// ---------------------------------------------------------------------------
template<bool VERT, int ROW_LEN>
__device__ __forceinline__
void async_load_row(const float* __restrict__ out, size_t nbase, int r,
                    float* ldsX, int tid)
{
    if (VERT) {
        const int CHUNKS = CCH * ROW_LEN / 4;             // 16B chunks
        for (int k = tid; k < CHUNKS; k += 512) {
            int c = k / (ROW_LEN / 4);
            int q = k % (ROW_LEN / 4);
            const float* g = out + nbase + (size_t)c * HH * WW + (size_t)r * WW + q * 4;
            unsigned l = (unsigned)(size_t)&ldsX[k * 4];
            asm volatile("global_load_async_to_lds_b128 %0, %1, off"
                         :: "v"(l), "v"((unsigned long long)(size_t)g) : "memory");
        }
    } else {
        for (int i = tid; i < CCH * ROW_LEN; i += 512) {  // one dword per element
            int c = i / ROW_LEN;
            int h = i % ROW_LEN;
            const float* g = out + nbase + (size_t)c * HH * WW + (size_t)h * WW + r;
            unsigned l = (unsigned)(size_t)&ldsX[i];
            asm volatile("global_load_async_to_lds_b32 %0, %1, off"
                         :: "v"(l), "v"((unsigned long long)(size_t)g) : "memory");
        }
    }
}

__device__ __forceinline__ void wait_async_all()
{
    asm volatile("s_wait_asynccnt 0x0" ::: "memory");
}

// ---------------------------------------------------------------------------
// One reverse sweep pass, in-place on `out`.
//   VERT:  scan over rows r (H=128), conv along w (len 192)
//   !VERT: scan over cols r (W=192), conv along h (len 128)
// out[r] = x[r] + relu(conv1d(out[r+1], w)),  r = len-2 .. 0
// One workgroup per batch item; 512 threads = 16 wave32s.
// ---------------------------------------------------------------------------
template<bool VERT>
__global__ __launch_bounds__(512)
void sweep_pass_kernel(float* __restrict__ out, const __bf16* __restrict__ wb)
{
    constexpr int ROW_LEN  = VERT ? WW : HH;   // conv-axis length
    constexpr int SCAN_LEN = VERT ? HH : WW;   // recurrence length
    constexpr int NT       = ROW_LEN / 16;     // N tiles (12 or 8)
    constexpr int NTW      = NT / 2;           // N tiles per wave (6 or 4)
    constexpr int NU       = KK * 4;           // 36 K-chunks of 32 (9 taps x 4)

    __shared__ __bf16 ldsC[LDS_ROWS * LDS_STRIDE];  // bf16 carry, [pos+PAD][ic]
    __shared__ float  ldsX[CCH * WW];               // fp32 row buffer, [c][pos]

    const int tid  = threadIdx.x;
    const int lane = tid & 31;
    const int lrow = lane & 15;   // fragment row / N column
    const int lhi  = lane >> 4;   // half select
    const int wave = tid >> 5;    // 0..15
    const int mt   = wave >> 1;   // M tile 0..7 (oc block)
    const int nt0  = (wave & 1) * NTW;

    const size_t nbase = (size_t)blockIdx.x * CCH * HH * WW;

    // Fragment loaders. u = t*4 + kc  (tap t, 32-wide ic chunk kc).
    const __bf16* wbase = wb + (size_t)(mt * 16 + lrow) * CCH + lhi * 16;
    auto ldA = [&](int u) -> v16bf {
        int t = u >> 2, kc = u & 3;
        return *(const v16bf*)(wbase + (size_t)t * CCH * CCH + kc * 32);
    };
    // per-thread LDS byte base for B fragments (j/t/kc added as offsets)
    const __bf16* bbase = ldsC + (nt0 * 16 + lrow) * LDS_STRIDE + lhi * 16;
    auto ldB = [&](int u, int j) -> v16bf {
        int t = u >> 2, kc = u & 3;
        return *(const v16bf*)(bbase + (j * 16 + t) * LDS_STRIDE + kc * 32);
    };

    // Zero carry buffer (halo rows and padding stay zero forever)
    for (int i = tid; i < LDS_ROWS * LDS_STRIDE; i += 512) ldsC[i] = (__bf16)0.f;

    // Initial carry = last scan row of `out` (unchanged by this pass)
    async_load_row<VERT, ROW_LEN>(out, nbase, SCAN_LEN - 1, ldsX, tid);
    wait_async_all();
    __syncthreads();
    for (int i = tid; i < CCH * ROW_LEN; i += 512) {
        int ic  = i / ROW_LEN;
        int pos = i % ROW_LEN;
        ldsC[(pos + PAD) * LDS_STRIDE + ic] = (__bf16)ldsX[i];
    }
    __syncthreads();

    for (int r = SCAN_LEN - 2; r >= 0; --r) {
        // Kick off async fetch of this step's x row; lands during the GEMM.
        async_load_row<VERT, ROW_LEN>(out, nbase, r, ldsX, tid);

        // ---- GEMM: acc[oc][pos] = sum_{t,ic} W[t][oc][ic] * carry[ic][pos+t-PAD] ----
        // Per K-chunk: batch-issue all NTW B-fragment loads (12 ds_load_b128),
        // then run the WMMAs; loads for later tiles stay in flight across the
        // earlier WMMAs. A fragment is prefetched one chunk ahead (global/L2).
        v8f acc[NTW] = {};
        v16bf a = ldA(0);
        #pragma unroll 2
        for (int u = 0; u < NU; ++u) {
            v16bf b[NTW];
            #pragma unroll
            for (int j = 0; j < NTW; ++j) b[j] = ldB(u, j);
            v16bf an = ldA((u + 1 < NU) ? u + 1 : u);   // dummy re-load on last
            #pragma unroll
            for (int j = 0; j < NTW; ++j) {
                acc[j] = __builtin_amdgcn_wmma_f32_16x16x32_bf16(
                    false, a, false, b[j], (short)0, acc[j], false, false);
            }
            a = an;
        }

        wait_async_all();    // x row landed in ldsX
        __syncthreads();     // ... for all waves; old-carry reads complete

        // ---- epilogue: v = x + relu(acc); new carry -> ldsC; result -> ldsX/out ----
        #pragma unroll
        for (int j = 0; j < NTW; ++j) {
            int pos = (nt0 + j) * 16 + lrow;          // N column = position
            v8bf pk;
            #pragma unroll
            for (int i = 0; i < 8; ++i) {
                int oc = mt * 16 + lhi * 8 + i;       // C/D VGPR i -> row M = lhi*8+i
                float v = ldsX[oc * ROW_LEN + pos] + fmaxf(acc[j][i], 0.f);
                pk[i] = (__bf16)v;
                if (VERT) {
                    ldsX[oc * ROW_LEN + pos] = v;     // own-element RMW, no race
                } else {
                    out[nbase + (size_t)oc * HH * WW + (size_t)pos * WW + r] = v;
                }
            }
            // 8 consecutive ic values -> one 16B LDS store of the new carry
            *(v8bf*)&ldsC[(pos + PAD) * LDS_STRIDE + mt * 16 + lhi * 8] = pk;
        }

        if (VERT) {
            __syncthreads();  // row complete in ldsX
            // Coalesced b128 stores of the finished row
            const int CHUNKS = CCH * ROW_LEN / 4;
            for (int k = tid; k < CHUNKS; k += 512) {
                int c = k / (ROW_LEN / 4);
                int q = k % (ROW_LEN / 4);
                *(v4f*)&out[nbase + (size_t)c * HH * WW + (size_t)r * WW + q * 4] =
                    *(const v4f*)&ldsX[k * 4];
            }
        }

        __syncthreads();      // ldsX free + new carry visible before next step
    }
}

// ---------------------------------------------------------------------------
extern "C" void kernel_launch(void* const* d_in, const int* in_sizes, int n_in,
                              void* d_out, int out_size, void* d_ws, size_t ws_size,
                              hipStream_t stream)
{
    (void)in_sizes; (void)n_in; (void)out_size; (void)ws_size;

    const float* x    = (const float*)d_in[0];
    const float* w_ud = (const float*)d_in[1];
    const float* w_du = (const float*)d_in[2];
    const float* w_lr = (const float*)d_in[3];
    const float* w_rl = (const float*)d_in[4];
    float*  out = (float*)d_out;
    __bf16* wb  = (__bf16*)d_ws;   // 4 * 9*128*128 bf16 = 1.2 MB

    const size_t xbytes = (size_t)4 * CCH * HH * WW * sizeof(float);

    // out <- x (all passes then run in place on out)
    hipMemcpyAsync(out, x, xbytes, hipMemcpyDeviceToDevice, stream);

    convert_weights_kernel<<<128, 512, 0, stream>>>(w_ud, w_du, w_lr, w_rl, wb);

    const size_t wstep = (size_t)KK * CCH * CCH;
    sweep_pass_kernel<true ><<<4, 512, 0, stream>>>(out, wb + 0 * wstep); // ud (vertical)
    sweep_pass_kernel<true ><<<4, 512, 0, stream>>>(out, wb + 1 * wstep); // du (vertical)
    sweep_pass_kernel<false><<<4, 512, 0, stream>>>(out, wb + 2 * wstep); // lr (horizontal)
    sweep_pass_kernel<false><<<4, 512, 0, stream>>>(out, wb + 3 * wstep); // rl (horizontal)
}
